// SimpleQuantizer_41472204210906
// MI455X (gfx1250) — matrix operand order repaired and checked by reading, not compiled, not run on hardware
//
#include <hip/hip_runtime.h>
#include <hip/hip_bf16.h>
#include <stdint.h>

typedef __attribute__((ext_vector_type(16))) __bf16 v16bf;
typedef __attribute__((ext_vector_type(8)))  float  v8f;
typedef __attribute__((ext_vector_type(4)))  float  v4f;

#define BB    8
#define NN    4096
#define CC    256
#define KK    1024
#define ROWS  (BB * NN)          // 32768
#define NUMEL (ROWS * CC)        // 8388608
#define NTILE (KK / 16)          // 64

// ---------- bf16 split helpers (truncation split: x = hi + lo, |err| ~ 2^-16 rel) ----------
__device__ __forceinline__ unsigned short bf16_trunc(float f) {
    union { float f; uint32_t u; } c; c.f = f;
    return (unsigned short)(c.u >> 16);
}
__device__ __forceinline__ float bf16_up(unsigned short h) {
    union { float f; uint32_t u; } c; c.u = ((uint32_t)h) << 16;
    return c.f;
}

// ---------- Kernel 1: codebook -> bf16 hi/lo + exact fp32 ||e_k||^2 ----------
__global__ void __launch_bounds__(256)
prep_codebook(const float* __restrict__ cb,
              unsigned short* __restrict__ cb_hi,
              unsigned short* __restrict__ cb_lo,
              float* __restrict__ e2) {
    const int k = blockIdx.x;
    const int c = threadIdx.x;
    float v = cb[(size_t)k * CC + c];
    unsigned short hi = bf16_trunc(v);
    float rem = v - bf16_up(hi);
    unsigned short lo = bf16_trunc(rem);
    cb_hi[(size_t)k * CC + c] = hi;
    cb_lo[(size_t)k * CC + c] = lo;

    float s = v * v;
    __shared__ float red[8];
    #pragma unroll
    for (int m = 16; m; m >>= 1) s += __shfl_xor(s, m, 32);
    const int lane = c & 31, w = c >> 5;
    if (lane == 0) red[w] = s;
    __syncthreads();
    if (c == 0) {
        float t = 0.f;
        #pragma unroll
        for (int i = 0; i < 8; ++i) t += red[i];
        e2[k] = t;
    }
}

// ---------- Kernel 2: fused WMMA distance GEMM + row argmin ----------
// Wave handles 16 rows of x. A fragments (hi/lo) held in VGPRs for the whole
// K=1024 sweep. dist(row,k) = e2[k] - 2 * x.e_k  (||x||^2 dropped: constant per row).
__global__ void __launch_bounds__(256)
vq_argmin_wmma(const float* __restrict__ x,
               const unsigned short* __restrict__ cb_hi,
               const unsigned short* __restrict__ cb_lo,
               const float* __restrict__ e2,
               int* __restrict__ out_idx) {
    const int lane = threadIdx.x & 31;
    const int wave = threadIdx.x >> 5;
    const int row_base = blockIdx.x * 128 + wave * 16;

    const int half = lane >> 4;        // 0: lanes 0-15, 1: lanes 16-31
    const int col  = lane & 15;        // A: row within tile; B/D: column (codebook id mod 16)
    const int koff = half * 16;        // A-fragment K offset: 16 contiguous K per lane half

    // ---- load 16x256 fp32 x tile once; split into bf16 hi/lo A fragments ----
    v16bf a_hi[8], a_lo[8];
    const float* xrow = x + (size_t)(row_base + col) * CC;
    #pragma unroll
    for (int c = 0; c < 8; ++c) {
        const float* p = xrow + c * 32 + koff;
        v4f f0 = *(const v4f*)(p + 0);
        v4f f1 = *(const v4f*)(p + 4);
        v4f f2 = *(const v4f*)(p + 8);    // elements 8..15 live at K = koff+8..koff+15
        v4f f3 = *(const v4f*)(p + 12);
        float f[16];
        #pragma unroll
        for (int j = 0; j < 4; ++j) { f[j] = f0[j]; f[4+j] = f1[j]; f[8+j] = f2[j]; f[12+j] = f3[j]; }
        union { v16bf v; unsigned short s[16]; } hi, lo;
        #pragma unroll
        for (int j = 0; j < 16; ++j) {
            unsigned short h = bf16_trunc(f[j]);
            hi.s[j] = h;
            lo.s[j] = bf16_trunc(f[j] - bf16_up(h));
        }
        a_hi[c] = hi.v;
        a_lo[c] = lo.v;
    }

    float minv[8];
    int   mini[8];
    #pragma unroll
    for (int i = 0; i < 8; ++i) { minv[i] = 3.4e38f; mini[i] = 0; }

    // B fragment, contiguous K order matching A: lane(col) holds 16 contiguous K values
    // starting at half*16 within each K=32 chunk, from codebook row (t*16+col).
    const int boff = half * 16;
    for (int t = 0; t < NTILE; ++t) {
        const int krow = t * 16 + col;
        const float e2v = e2[krow];
        const unsigned short* bh = cb_hi + (size_t)krow * CC + boff;
        const unsigned short* bl = cb_lo + (size_t)krow * CC + boff;

        v8f acc = {};
        #pragma unroll
        for (int c = 0; c < 8; ++c) {
            v16bf bhi = *(const v16bf*)(bh + c * 32);
            v16bf blo = *(const v16bf*)(bl + c * 32);
            acc = __builtin_amdgcn_wmma_f32_16x16x32_bf16(false, a_hi[c], false, bhi,
                                                          (short)0, acc, false, false);
            acc = __builtin_amdgcn_wmma_f32_16x16x32_bf16(false, a_hi[c], false, blo,
                                                          (short)0, acc, false, false);
            acc = __builtin_amdgcn_wmma_f32_16x16x32_bf16(false, a_lo[c], false, bhi,
                                                          (short)0, acc, false, false);
        }
        // D element i in this lane: row = row_base + i + 8*half, column krow (same for all i).
        #pragma unroll
        for (int i = 0; i < 8; ++i) {
            float d = e2v - 2.0f * acc[i];
            if (d < minv[i]) { minv[i] = d; mini[i] = krow; }
        }
    }

    // min+argmin across the 16 lanes of each half (first-occurrence tie-break: lower k wins)
    #pragma unroll
    for (int i = 0; i < 8; ++i) {
        float v = minv[i]; int id = mini[i];
        #pragma unroll
        for (int m = 1; m < 16; m <<= 1) {
            float ov = __shfl_xor(v, m, 32);
            int   oi = __shfl_xor(id, m, 32);
            if (ov < v || (ov == v && oi < id)) { v = ov; id = oi; }
        }
        minv[i] = v; mini[i] = id;
    }
    if (col == 0) {
        #pragma unroll
        for (int i = 0; i < 8; ++i)
            out_idx[row_base + i + 8 * half] = mini[i];
    }
}

// ---------- Kernel 3: gather fp32 codebook rows, stream x, per-row loss partials ----------
__global__ void __launch_bounds__(256)
gather_loss(const float* __restrict__ x, const float* __restrict__ cb,
            const int* __restrict__ idx, float* __restrict__ qout,
            float* __restrict__ idx_out, float* __restrict__ partial) {
    const int row = blockIdx.x;
    const int c   = threadIdx.x;
    const int k   = idx[row];
    float q  = cb[(size_t)k * CC + c];
    float xv = x[(size_t)row * CC + c];
    qout[(size_t)row * CC + c] = q;          // quant_out == quant (straight-through)
    float d = q - xv;
    float s = d * d;
    __shared__ float red[8];
    #pragma unroll
    for (int m = 16; m; m >>= 1) s += __shfl_xor(s, m, 32);
    const int lane = c & 31, w = c >> 5;
    if (lane == 0) red[w] = s;
    __syncthreads();
    if (c == 0) {
        float t = 0.f;
        #pragma unroll
        for (int i = 0; i < 8; ++i) t += red[i];
        partial[row] = t;
        idx_out[row] = (float)k;             // indices emitted as float (d_out is float*)
    }
}

// ---------- Kernel 4: deterministic final loss reduction ----------
__global__ void __launch_bounds__(256)
final_reduce(const float* __restrict__ partial, float* __restrict__ loss_out) {
    float s = 0.f;
    for (int i = threadIdx.x; i < ROWS; i += 256) s += partial[i];
    __shared__ float red[8];
    #pragma unroll
    for (int m = 16; m; m >>= 1) s += __shfl_xor(s, m, 32);
    const int lane = threadIdx.x & 31, w = threadIdx.x >> 5;
    if (lane == 0) red[w] = s;
    __syncthreads();
    if (threadIdx.x == 0) {
        float t = 0.f;
        #pragma unroll
        for (int i = 0; i < 8; ++i) t += red[i];
        // codebook_loss + BETA*commitment_loss = (1 + 0.25) * mean((quant-x)^2)
        loss_out[0] = t * (1.25f / (float)NUMEL);
    }
}

extern "C" void kernel_launch(void* const* d_in, const int* in_sizes, int n_in,
                              void* d_out, int out_size, void* d_ws, size_t ws_size,
                              hipStream_t stream) {
    const float* x  = (const float*)d_in[0];   // [8,4096,256] fp32
    const float* cb = (const float*)d_in[1];   // [1024,256]   fp32
    float* out = (float*)d_out;
    // d_out layout (return order): quant_out[NUMEL], loss[1], indices[ROWS]
    float* quant_out = out;
    float* loss_out  = out + NUMEL;
    float* idx_out   = out + NUMEL + 1;

    // workspace carve-up (~1.3 MB total)
    char* w = (char*)d_ws;
    unsigned short* cb_hi = (unsigned short*)(w);                 // 512 KB
    unsigned short* cb_lo = (unsigned short*)(w + (512u << 10));  // 512 KB
    float* e2      = (float*)(w + (1024u << 10));                 //   4 KB
    int*   idx     = (int*)  (w + (1028u << 10));                 // 128 KB
    float* partial = (float*)(w + (1156u << 10));                 // 128 KB

    prep_codebook <<<KK,        256, 0, stream>>>(cb, cb_hi, cb_lo, e2);
    vq_argmin_wmma<<<ROWS / 128, 256, 0, stream>>>(x, cb_hi, cb_lo, e2, idx);
    gather_loss   <<<ROWS,      256, 0, stream>>>(x, cb, idx, quant_out, idx_out, partial);
    final_reduce  <<<1,         256, 0, stream>>>(partial, loss_out);
}
